// Model_42803644072529
// MI455X (gfx1250) — matrix operand order repaired and checked by reading, not compiled, not run on hardware
//
#include <hip/hip_runtime.h>

#define DH 128
#define WT_STRIDE 136   // padded ushort stride: keeps 16B alignment, rotates LDS banks

typedef __attribute__((ext_vector_type(16))) __bf16 v16bf;
typedef __attribute__((ext_vector_type(8)))  float  v8f;

union BFrag {
  v16bf  v;
  __bf16 h[16];
  uint4  q[2];
};

__device__ __forceinline__ unsigned short f2bf(float f) {
  unsigned u = __float_as_uint(f);
  unsigned r = ((u >> 16) & 1u) + 0x7FFFu;   // round-to-nearest-even
  return (unsigned short)((u + r) >> 16);
}

// order-preserving float<->uint encoding for atomicMax on signed floats
__device__ __forceinline__ unsigned encf(float f) {
  unsigned u = __float_as_uint(f);
  return (u & 0x80000000u) ? ~u : (u | 0x80000000u);
}
__device__ __forceinline__ float decf(unsigned e) {
  unsigned u = (e & 0x80000000u) ? (e & 0x7FFFFFFFu) : ~e;
  return __uint_as_float(u);
}

// ---------------- GEMM: H[n x 128] = X[n x 128] @ W[128 x 128], bf16 WMMA ----
__global__ __launch_bounds__(128) void gat_gemm(const float* __restrict__ X,
                                                const float* __restrict__ Wm,
                                                float* __restrict__ H, int nrows) {
  __shared__ unsigned short Wt[DH * WT_STRIDE];   // Wt[n][k] = bf16(W[k][n])
  const int tid = threadIdx.x;
  for (int k = 0; k < DH; ++k)                    // coalesced global reads of W row k
    Wt[tid * WT_STRIDE + k] = f2bf(Wm[(size_t)k * DH + tid]);
  __syncthreads();

  const int wave  = tid >> 5;
  const int lane  = tid & 31;
  const int khalf = lane >> 4;      // 0: lanes 0-15, 1: lanes 16-31
  const int l16   = lane & 15;
  const int rowbase = blockIdx.x * 64 + wave * 16;

  int rA = rowbase + l16;
  if (rA > nrows - 1) rA = nrows - 1;             // clamp (no EXEC divergence)
  const float* __restrict__ xrow = X + (size_t)rA * DH;

  v8f acc[8];
#pragma unroll
  for (int i = 0; i < 8; ++i) acc[i] = (v8f){0.f,0.f,0.f,0.f,0.f,0.f,0.f,0.f};

#pragma unroll
  for (int kt = 0; kt < 4; ++kt) {
    // A fragment: 16x32 bf16 per ISA layout (lane half selects K runs)
    const float* xa = xrow + kt * 32 + khalf * 8;
    float4 a0 = *(const float4*)(xa);
    float4 a1 = *(const float4*)(xa + 4);
    float4 a2 = *(const float4*)(xa + 16);
    float4 a3 = *(const float4*)(xa + 20);
    BFrag A;
    A.h[0]=(__bf16)a0.x; A.h[1]=(__bf16)a0.y; A.h[2]=(__bf16)a0.z; A.h[3]=(__bf16)a0.w;
    A.h[4]=(__bf16)a1.x; A.h[5]=(__bf16)a1.y; A.h[6]=(__bf16)a1.z; A.h[7]=(__bf16)a1.w;
    A.h[8]=(__bf16)a2.x; A.h[9]=(__bf16)a2.y; A.h[10]=(__bf16)a2.z; A.h[11]=(__bf16)a2.w;
    A.h[12]=(__bf16)a3.x; A.h[13]=(__bf16)a3.y; A.h[14]=(__bf16)a3.z; A.h[15]=(__bf16)a3.w;

#pragma unroll
    for (int nt = 0; nt < 8; ++nt) {
      // B fragment: lane holds column n = nt*16+l16, K runs 8*half..+7 and 16+8*half..+7
      const unsigned short* wr =
          &Wt[(nt * 16 + l16) * WT_STRIDE + kt * 32 + khalf * 8];
      BFrag B;
      B.q[0] = *(const uint4*)(wr);
      B.q[1] = *(const uint4*)(wr + 16);
      acc[nt] = __builtin_amdgcn_wmma_f32_16x16x32_bf16(
          false, A.v, false, B.v, (short)0, acc[nt], false, false);
    }
  }

#pragma unroll
  for (int nt = 0; nt < 8; ++nt) {
#pragma unroll
    for (int j = 0; j < 8; ++j) {
      const int gr = rowbase + j + 8 * khalf;     // C/D layout: VGPR j -> M=j(+8)
      if (gr < nrows) H[(size_t)gr * DH + nt * 16 + l16] = acc[nt][j];
    }
  }
}

// ---------------- per-node attention logits: al_s = h.a_src, al_d = h.a_dst --
__global__ __launch_bounds__(256) void node_alpha(const float* __restrict__ H,
                                                  const float* __restrict__ as,
                                                  const float* __restrict__ ad,
                                                  float* __restrict__ als,
                                                  float* __restrict__ ald, int n) {
  const int w = (int)((blockIdx.x * (unsigned)blockDim.x + threadIdx.x) >> 5);
  const int lane = threadIdx.x & 31;
  if (w >= n) return;
  float4 hv = *(const float4*)(H + (size_t)w * DH + lane * 4);
  float4 sv = *(const float4*)(as + lane * 4);
  float4 dv = *(const float4*)(ad + lane * 4);
  float s = hv.x*sv.x + hv.y*sv.y + hv.z*sv.z + hv.w*sv.w;
  float d = hv.x*dv.x + hv.y*dv.y + hv.z*dv.z + hv.w*dv.w;
  for (int off = 16; off; off >>= 1) {
    s += __shfl_down(s, off);
    d += __shfl_down(d, off);
  }
  if (lane == 0) { als[w] = s; ald[w] = d; }
}

// ---------------- per-layer scratch init ------------------------------------
__global__ void node_init(float* den, unsigned* mx, int n) {
  int i = blockIdx.x * blockDim.x + threadIdx.x;
  if (i < n) { den[i] = 0.f; mx[i] = 0u; }   // 0u encodes < encf(-inf)
}
__global__ void out_init(float* O, const float* __restrict__ bias, int total) {
  int i = blockIdx.x * blockDim.x + threadIdx.x;
  if (i < total) O[i] = bias[i & (DH - 1)];
}

// ---------------- edge pass 1: segment max of leaky-relu logits --------------
__global__ void edge_max(const int* __restrict__ src, const int* __restrict__ dst,
                         const float* __restrict__ als, const float* __restrict__ ald,
                         unsigned* __restrict__ mxe, int nE, int nN) {
  int i = blockIdx.x * blockDim.x + threadIdx.x;
  if (i >= nE + nN) return;
  int s, d;
  if (i < nE) { s = src[i]; d = dst[i]; } else { s = d = i - nE; }
  float v = als[s] + ald[d];
  v = v > 0.f ? v : 0.2f * v;
  atomicMax(mxe + d, encf(v));
}

// ---------------- edge pass 2: exp + segment sum -----------------------------
__global__ void edge_exp(const int* __restrict__ src, const int* __restrict__ dst,
                         const float* __restrict__ als, const float* __restrict__ ald,
                         const unsigned* __restrict__ mxe, float* __restrict__ den,
                         float* __restrict__ ex, int nE, int nN) {
  int i = blockIdx.x * blockDim.x + threadIdx.x;
  if (i >= nE + nN) return;
  int s, d;
  if (i < nE) { s = src[i]; d = dst[i]; } else { s = d = i - nE; }
  float v = als[s] + ald[d];
  v = v > 0.f ? v : 0.2f * v;
  float e = __expf(v - decf(mxe[d]));
  ex[i] = e;
  atomicAdd(den + d, e);
}

// ---------------- edge pass 3: alpha-weighted scatter (wave per edge) --------
__global__ __launch_bounds__(256) void edge_agg(const int* __restrict__ src,
                                                const int* __restrict__ dst,
                                                const float* __restrict__ H,
                                                const float* __restrict__ ex,
                                                const float* __restrict__ den,
                                                float* __restrict__ O, int nE, int nN) {
  const long long w = ((long long)blockIdx.x * blockDim.x + threadIdx.x) >> 5;
  const int lane = threadIdx.x & 31;
  if (w >= (long long)nE + nN) return;
  int s, d;
  if (w < nE) { s = src[w]; d = dst[w]; } else { s = d = (int)(w - nE); }
  __builtin_prefetch(H + (size_t)s * DH + lane * 4, 0, 0);  // global_prefetch
  const float alpha = ex[w] / den[d];
  float4 hv = *(const float4*)(H + (size_t)s * DH + lane * 4);
  float* o = O + (size_t)d * DH + lane * 4;
  atomicAdd(o + 0, alpha * hv.x);
  atomicAdd(o + 1, alpha * hv.y);
  atomicAdd(o + 2, alpha * hv.z);
  atomicAdd(o + 3, alpha * hv.w);
}

// ---------------- global add pool -------------------------------------------
__global__ void pool_zero(float* g) { g[blockIdx.x * blockDim.x + threadIdx.x] = 0.f; }

__global__ __launch_bounds__(256) void pool_scatter(const float* __restrict__ X,
                                                    const int* __restrict__ batch,
                                                    float* __restrict__ g, int n) {
  const int w = (int)((blockIdx.x * (unsigned)blockDim.x + threadIdx.x) >> 5);
  const int lane = threadIdx.x & 31;
  if (w >= n) return;
  const int b = batch[w];
  float4 xv = *(const float4*)(X + (size_t)w * DH + lane * 4);
  float* gp = g + (size_t)b * DH + lane * 4;
  atomicAdd(gp + 0, xv.x);
  atomicAdd(gp + 1, xv.y);
  atomicAdd(gp + 2, xv.z);
  atomicAdd(gp + 3, xv.w);
}

// ---------------- readout MLP: relu(g@W1+b1)@W2+b2 ---------------------------
__global__ __launch_bounds__(64) void readout(const float* __restrict__ g,
                                              const float* __restrict__ W1,
                                              const float* __restrict__ b1,
                                              const float* __restrict__ W2,
                                              const float* __restrict__ b2,
                                              float* __restrict__ out) {
  __shared__ float hid[64];
  const int gr = blockIdx.x, t = threadIdx.x;
  float s = b1[t];
  for (int k = 0; k < DH; ++k) s += g[gr * DH + k] * W1[k * 64 + t];
  hid[t] = fmaxf(s, 0.f);
  __syncthreads();
  if (t < 10) {
    float o = b2[t];
    for (int k = 0; k < 64; ++k) o += hid[k] * W2[k * 10 + t];
    out[gr * 10 + t] = o;
  }
}

extern "C" void kernel_launch(void* const* d_in, const int* in_sizes, int n_in,
                              void* d_out, int out_size, void* d_ws, size_t ws_size,
                              hipStream_t stream) {
  const float* x     = (const float*)d_in[0];
  const int*   ei    = (const int*)d_in[1];
  const int*   batch = (const int*)d_in[2];
  const float* W     = (const float*)d_in[3];
  const float* a_src = (const float*)d_in[4];
  const float* a_dst = (const float*)d_in[5];
  const float* bias  = (const float*)d_in[6];
  const float* W1    = (const float*)d_in[7];
  const float* b1    = (const float*)d_in[8];
  const float* W2    = (const float*)d_in[9];
  const float* b2    = (const float*)d_in[10];
  float* out = (float*)d_out;

  const int N = in_sizes[0] / DH;      // 100000
  const int E = in_sizes[1] / 2;       // 1600000
  const int ET = E + N;                // edges + self-loops
  const int* srcp = ei;
  const int* dstp = ei + E;

  // workspace carve-up
  float* ws  = (float*)d_ws;
  float* Hh  = ws;                          // N*128
  float* O1  = Hh + (size_t)N * DH;         // N*128
  float* O2  = O1 + (size_t)N * DH;         // N*128
  float* als = O2 + (size_t)N * DH;         // N
  float* ald = als + N;                     // N
  float* den = ald + N;                     // N
  unsigned* mxe = (unsigned*)(den + N);     // N
  float* ex  = (float*)(mxe + N);           // E+N
  float* g   = ex + ET;                     // 128*128

  const float* xin = x;
  float* obuf[3] = {O1, O2, O1};

  for (int l = 0; l < 3; ++l) {
    float* O = obuf[l];
    gat_gemm<<<(N + 63) / 64, 128, 0, stream>>>(xin, W + (size_t)l * DH * DH, Hh, N);
    node_alpha<<<(N * 32 + 255) / 256, 256, 0, stream>>>(
        Hh, a_src + l * DH, a_dst + l * DH, als, ald, N);
    node_init<<<(N + 255) / 256, 256, 0, stream>>>(den, mxe, N);
    out_init<<<(N * DH + 255) / 256, 256, 0, stream>>>(O, bias + l * DH, N * DH);
    edge_max<<<(ET + 255) / 256, 256, 0, stream>>>(srcp, dstp, als, ald, mxe, E, N);
    edge_exp<<<(ET + 255) / 256, 256, 0, stream>>>(srcp, dstp, als, ald, mxe, den, ex, E, N);
    edge_agg<<<(int)(((long long)ET * 32 + 255) / 256), 256, 0, stream>>>(
        srcp, dstp, Hh, ex, den, O, E, N);
    xin = O;
  }

  pool_zero<<<64, 256, 0, stream>>>(g);                 // 128*128 = 64*256
  pool_scatter<<<(N * 32 + 255) / 256, 256, 0, stream>>>(xin, batch, g, N);
  readout<<<128, 64, 0, stream>>>(g, W1, b1, W2, b2, out);
}
